// PhonemeEncoder_29394756173825
// MI455X (gfx1250) — matrix operand order repaired and checked by reading, not compiled, not run on hardware
//
#include <hip/hip_runtime.h>
#include <cstdint>

#define B_   32
#define P_   1024
#define PP_  1026          // padded positions (1 halo row each side)
#define EH_  512           // E == H == 512
#define KSTEPS_ 48         // 3*512 / 32

typedef __attribute__((ext_vector_type(16))) __bf16 v16bf;
typedef __attribute__((ext_vector_type(8)))  __bf16 v8bf;
typedef __attribute__((ext_vector_type(8)))  float  v8f;
typedef __attribute__((ext_vector_type(4)))  unsigned int u32x4;
typedef __attribute__((ext_vector_type(8)))  int    i32x8;
typedef __attribute__((ext_vector_type(4)))  int    i32x4;

union AFrag { v16bf v; v8bf h[2]; };

// ---------------------------------------------------------------------------
// Embedding gather + fp32->bf16, position-major padded layout X0[b][1026][512]
// ---------------------------------------------------------------------------
__global__ __launch_bounds__(256) void embed_kernel(const int* __restrict__ ids,
                                                    const float* __restrict__ emb,
                                                    __bf16* __restrict__ X0) {
  const int b = blockIdx.y;
  const int row = blockIdx.x;              // 0..1025
  const int t = threadIdx.x;               // 256 threads, 2 elems each
  __bf16* dst = X0 + ((size_t)b * PP_ + row) * EH_;
  if (row == 0 || row == PP_ - 1) {
    dst[t] = (__bf16)0.0f; dst[t + 256] = (__bf16)0.0f;
    return;
  }
  const int id = ids[b * P_ + (row - 1)];
  const float* src = emb + (size_t)id * EH_;
  dst[t]       = (__bf16)src[t];
  dst[t + 256] = (__bf16)src[t + 256];
}

// Zero halo rows of the padded Y1 buffer
__global__ __launch_bounds__(256) void halo_zero_kernel(__bf16* __restrict__ Y) {
  const int b = blockIdx.y;
  const int row = blockIdx.x ? (PP_ - 1) : 0;
  __bf16* dst = Y + ((size_t)b * PP_ + row) * EH_;
  dst[threadIdx.x] = (__bf16)0.0f;
  dst[threadIdx.x + 256] = (__bf16)0.0f;
}

// ---------------------------------------------------------------------------
// Pack conv weights (H,E,3) fp32 into WMMA B-fragment-linear bf16 layout:
// flat[((tn*48 + kk)*32 + lane)*16 + j]  with  n = tn*16 + (lane&15),
// K = kk*32 + (lane>>4)*16 + j,  k = K/512, e = K%512,  value = w[h][e][k]
// ---------------------------------------------------------------------------
__global__ __launch_bounds__(256) void prepack_w_kernel(const float* __restrict__ w,
                                                        __bf16* __restrict__ Wp) {
  const size_t idx = (size_t)blockIdx.x * 256 + threadIdx.x;   // < 512*1536
  const int j    = (int)(idx & 15);
  const int lane = (int)((idx >> 4) & 31);
  const int kk   = (int)((idx >> 9) % KSTEPS_);
  const int tn   = (int)(idx / (KSTEPS_ * 512));
  const int h    = tn * 16 + (lane & 15);
  const int Kg   = kk * 32 + ((lane >> 4) * 16) + j;
  const int k    = Kg >> 9;
  const int e    = Kg & 511;
  Wp[idx] = (__bf16)w[((size_t)h * EH_ + e) * 3 + k];
}

// Fused per-channel scale/shift:  y = conv*s + t ;  s = gamma*rsqrt(var+eps),
// t = bias*s + beta - mean*s
__global__ __launch_bounds__(256) void bn_coeffs_kernel(const float* __restrict__ bias,
                                                        const float* __restrict__ gamma,
                                                        const float* __restrict__ beta,
                                                        const float* __restrict__ mean,
                                                        const float* __restrict__ var,
                                                        float* __restrict__ s,
                                                        float* __restrict__ t) {
  const int h = blockIdx.x * 256 + threadIdx.x;
  if (h < EH_) {
    const float sc = gamma[h] * rsqrtf(var[h] + 1e-5f);
    s[h] = sc;
    t[h] = beta[h] - mean[h] * sc + bias[h] * sc;
  }
}

// ---------------------------------------------------------------------------
// Conv(k=3,pad=1) + BN + ReLU as WMMA GEMM.
//   D[p][h] = sum_{k,e} X[b][p+k][e] * Wp[(k*512+e)][h]   (padded rows)
// Workgroup: 256 thr = 8 waves, tile 256(pos) x 64(h); wave tile 64x32.
// A tiles (32 channels x 258 rows) streamed global->LDS by the Tensor Data
// Mover, double buffered, synchronized with s_wait_tensorcnt + barriers.
// B fragments: one 32B contiguous load per lane from prepacked weights (L2).
// ---------------------------------------------------------------------------
template <bool STAGE2>
__global__ __launch_bounds__(256) void conv_bn_relu_kernel(
    const __bf16* __restrict__ Xin,   // [B][1026][512] bf16 padded
    const __bf16* __restrict__ Wp,    // fragment-packed [32][48][32][16]
    const float* __restrict__ sc, const float* __restrict__ sh,
    void* __restrict__ Yout) {
  __shared__ __bf16 Abuf[2][258 * 32];   // 2 x 16.5KB staging tiles

  const int lane = threadIdx.x & 31;
  const int wave = threadIdx.x >> 5;
  const int wm = wave & 3;               // wave M index (position)
  const int wn = wave >> 2;              // wave N index (channel)
  const int b = blockIdx.z;
  const int p0wg = blockIdx.x * 256;     // workgroup position base
  const int p0 = p0wg + wm * 64;
  const int h0 = blockIdx.y * 64 + wn * 32;
  const int tn0 = h0 >> 4;               // first 16-wide N tile index

  // --- TDM issue: 2D tile (tile_dim0=32 ch, tile_dim1=258 rows) -> LDS ---
  auto issue = [&](int kk, int bufsel) {
    const int e0 = (kk & 15) << 5;
    const uint64_t gaddr =
        (uint64_t)(uintptr_t)(Xin + (((size_t)b * PP_ + p0wg) * EH_ + e0));
    const uint32_t laddr = (uint32_t)(size_t)(&Abuf[bufsel][0]);
    u32x4 g0; i32x8 g1; i32x4 g2, g3; i32x8 g4;
    g0[0] = 1u;                                              // count=1
    g0[1] = laddr;                                           // lds_addr
    g0[2] = (uint32_t)gaddr;                                 // global_addr lo
    g0[3] = (uint32_t)((gaddr >> 32) & 0x1FFFFFFu) | (2u << 30);  // hi | type=2
    g1[0] = (int)(1u << 16);                                 // data_size=2B
    g1[1] = (int)((uint32_t)EH_ << 16);                      // tensor_dim0 lo
    g1[2] = (int)((uint32_t)PP_ << 16);                      // dim0 hi=0, dim1 lo
    g1[3] = (int)(32u << 16);                                // dim1 hi=0, tile_dim0=32
    g1[4] = (int)258u;                                       // tile_dim1=258
    g1[5] = (int)(uint32_t)EH_;                              // dim0_stride lo32
    g1[6] = (int)(((uint32_t)(PP_ * EH_) & 0xFFFFu) << 16);  // dim1_stride lo16
    g1[7] = (int)((uint32_t)(PP_ * EH_) >> 16);              // dim1_stride hi
    g2[0] = g2[1] = g2[2] = g2[3] = 0;
    g3[0] = g3[1] = g3[2] = g3[3] = 0;
    g4[0] = g4[1] = g4[2] = g4[3] = 0;
    g4[4] = g4[5] = g4[6] = g4[7] = 0;
    __builtin_amdgcn_tensor_load_to_lds(g0, g1, g2, g3, g4, 0);
  };

  v8f acc[4][2] = {};

  if (wave == 0) issue(0, 0);

  const int plocal_base = (p0 - p0wg) + (lane & 15);   // row-in-tile base
  const int kbase = (lane >> 4) * 16;                  // A-frag K sub-offset

  for (int kk = 0; kk < KSTEPS_; ++kk) {
    if (wave == 0) {
      if (kk + 1 < KSTEPS_) {
        issue(kk + 1, (kk + 1) & 1);
        __builtin_amdgcn_s_wait_tensorcnt(1);   // tile kk landed (in-order)
      } else {
        __builtin_amdgcn_s_wait_tensorcnt(0);
      }
    }
    __syncthreads();                            // tile kk visible to all waves

    const int kshift = kk >> 4;                 // conv tap 0..2 -> row offset
    const __bf16* Ab = &Abuf[kk & 1][0];

    // B fragments: 32B/lane contiguous from fragment-packed weights
    v16bf bf0 = *(const v16bf*)(Wp + (((size_t)(tn0 + 0) * KSTEPS_ + kk) * 32 + lane) * 16);
    v16bf bf1 = *(const v16bf*)(Wp + (((size_t)(tn0 + 1) * KSTEPS_ + kk) * 32 + lane) * 16);

#pragma unroll
    for (int mi = 0; mi < 4; ++mi) {
      const int row = plocal_base + mi * 16 + kshift;   // 0..257
      const __bf16* ap = Ab + row * 32 + kbase;
      AFrag a;
      a.h[0] = *(const v8bf*)(ap);        // K = kbase..kbase+7
      a.h[1] = *(const v8bf*)(ap + 8);    // K = kbase+8..kbase+15
      acc[mi][0] = __builtin_amdgcn_wmma_f32_16x16x32_bf16(
          false, a.v, false, bf0, (short)0, acc[mi][0], false, false);
      acc[mi][1] = __builtin_amdgcn_wmma_f32_16x16x32_bf16(
          false, a.v, false, bf1, (short)0, acc[mi][1], false, false);
    }
    __syncthreads();                            // done reading buf before reuse
  }

  // Epilogue: D layout -> m = mi*16 + r + 8*(lane>>4), n = lane&15
  const int nlo = lane & 15;
  const int msel = (lane >> 4) * 8;
#pragma unroll
  for (int ni = 0; ni < 2; ++ni) {
    const int h = h0 + ni * 16 + nlo;
    const float s = sc[h];
    const float t = sh[h];
#pragma unroll
    for (int mi = 0; mi < 4; ++mi) {
#pragma unroll
      for (int r = 0; r < 8; ++r) {
        const int p = p0 + mi * 16 + r + msel;
        float y = acc[mi][ni][r] * s + t;
        y = fmaxf(y, 0.0f);
        if (STAGE2) {
          ((float*)Yout)[((size_t)b * P_ + p) * EH_ + h] = y;
        } else {
          ((__bf16*)Yout)[((size_t)b * PP_ + 1 + p) * EH_ + h] = (__bf16)y;
        }
      }
    }
  }
}

// ---------------------------------------------------------------------------
// Inclusive scan of durations per batch row (Hillis-Steele in LDS)
// ---------------------------------------------------------------------------
__global__ __launch_bounds__(1024) void scan_kernel(const int* __restrict__ dur,
                                                    int* __restrict__ cum,
                                                    int* __restrict__ total) {
  __shared__ int shm[1024];
  const int b = blockIdx.x, t = threadIdx.x;
  shm[t] = dur[b * P_ + t];
  __syncthreads();
  for (int off = 1; off < 1024; off <<= 1) {
    const int add = (t >= off) ? shm[t - off] : 0;
    __syncthreads();
    shm[t] += add;
    __syncthreads();
  }
  cum[b * P_ + t] = shm[t];
  if (t == 1023) total[b] = shm[t];
}

// ---------------------------------------------------------------------------
// Upsample: binary-search frame->phoneme, gather rows of Y2[b][p][h],
// transpose through LDS so (B,H,F) writes are coalesced, mask f >= total.
// ---------------------------------------------------------------------------
__global__ __launch_bounds__(256) void upsample_kernel(const float* __restrict__ Y2,
                                                       const int* __restrict__ cum,
                                                       const int* __restrict__ total,
                                                       float* __restrict__ out, int F) {
  __shared__ float tile[64][65];
  __shared__ int sidx[64];
  __shared__ int stot;
  const int b = blockIdx.z;
  const int h0 = blockIdx.y * 64;
  const int f0 = blockIdx.x * 64;
  const int t = threadIdx.x;
  if (t == 0) stot = total[b];
  if (t < 64) {
    const int f = f0 + t;
    const int* c = cum + b * P_;
    int lo = 0, hi = P_;                 // searchsorted(..., side='right')
    while (lo < hi) {
      const int mid = (lo + hi) >> 1;
      if (c[mid] <= f) lo = mid + 1; else hi = mid;
    }
    sidx[t] = (lo < P_ - 1) ? lo : (P_ - 1);
  }
  __syncthreads();

  const int hl = t & 63, fr = t >> 6;
#pragma unroll 4
  for (int it = 0; it < 16; ++it) {
    const int fl = fr + it * 4;
    const int f = f0 + fl;
    float v = 0.0f;
    if (f < F && f < stot)
      v = Y2[((size_t)b * P_ + sidx[fl]) * EH_ + h0 + hl];
    tile[fl][hl] = v;
  }
  __syncthreads();

  const int fl2 = t & 63, hr = t >> 6;
#pragma unroll 4
  for (int it = 0; it < 16; ++it) {
    const int hl2 = hr + it * 4;
    const int f = f0 + fl2;
    if (f < F)
      out[((size_t)b * EH_ + h0 + hl2) * (size_t)F + f] = tile[fl2][hl2];
  }
}

// ---------------------------------------------------------------------------
extern "C" void kernel_launch(void* const* d_in, const int* in_sizes, int n_in,
                              void* d_out, int out_size, void* d_ws, size_t ws_size,
                              hipStream_t stream) {
  (void)in_sizes; (void)n_in; (void)ws_size;
  const int*   ids  = (const int*)d_in[0];
  const int*   dur  = (const int*)d_in[1];
  const float* emb  = (const float*)d_in[2];
  const float* w1   = (const float*)d_in[3];
  const float* b1   = (const float*)d_in[4];
  const float* g1   = (const float*)d_in[5];
  const float* be1  = (const float*)d_in[6];
  const float* m1   = (const float*)d_in[7];
  const float* v1   = (const float*)d_in[8];
  const float* w2   = (const float*)d_in[9];
  const float* b2   = (const float*)d_in[10];
  const float* g2   = (const float*)d_in[11];
  const float* be2  = (const float*)d_in[12];
  const float* m2   = (const float*)d_in[13];
  const float* v2   = (const float*)d_in[14];
  float* out = (float*)d_out;

  char* ws = (char*)d_ws;
  size_t off = 0;
  auto alloc = [&](size_t bytes) -> void* {
    off = (off + 255) & ~(size_t)255;
    void* p = ws + off;
    off += bytes;
    return p;
  };
  __bf16* X0   = (__bf16*)alloc((size_t)B_ * PP_ * EH_ * 2);
  __bf16* Y1   = (__bf16*)alloc((size_t)B_ * PP_ * EH_ * 2);
  float*  Y2   = (float*) alloc((size_t)B_ * P_  * EH_ * 4);
  __bf16* Wp1  = (__bf16*)alloc((size_t)EH_ * 1536 * 2);
  __bf16* Wp2  = (__bf16*)alloc((size_t)EH_ * 1536 * 2);
  float*  s1   = (float*) alloc(EH_ * 4);
  float*  t1   = (float*) alloc(EH_ * 4);
  float*  s2   = (float*) alloc(EH_ * 4);
  float*  t2   = (float*) alloc(EH_ * 4);
  int*    cum  = (int*)   alloc((size_t)B_ * P_ * 4);
  int*    tot  = (int*)   alloc(B_ * 4);

  const int F = out_size / (B_ * EH_);

  bn_coeffs_kernel<<<2, 256, 0, stream>>>(b1, g1, be1, m1, v1, s1, t1);
  bn_coeffs_kernel<<<2, 256, 0, stream>>>(b2, g2, be2, m2, v2, s2, t2);
  prepack_w_kernel<<<3072, 256, 0, stream>>>(w1, Wp1);
  prepack_w_kernel<<<3072, 256, 0, stream>>>(w2, Wp2);
  embed_kernel<<<dim3(PP_, B_), 256, 0, stream>>>(ids, emb, X0);
  halo_zero_kernel<<<dim3(2, B_), 256, 0, stream>>>(Y1);
  scan_kernel<<<B_, 1024, 0, stream>>>(dur, cum, tot);

  dim3 cgrid(P_ / 256, EH_ / 64, B_);
  conv_bn_relu_kernel<false><<<cgrid, 256, 0, stream>>>(X0, Wp1, s1, t1, (void*)Y1);
  conv_bn_relu_kernel<true ><<<cgrid, 256, 0, stream>>>(Y1, Wp2, s2, t2, (void*)Y2);

  if (F > 0) {
    dim3 ugrid((F + 63) / 64, EH_ / 64, B_);
    upsample_kernel<<<ugrid, 256, 0, stream>>>(Y2, cum, tot, out, F);
  }
}
